// CoDy_33784212750765
// MI455X (gfx1250) — compile-verified, matrix operand order checked by reading, not run on hardware
//
#include <hip/hip_runtime.h>

// CoDy forward on gfx1250 (MI455X). wave32; all dense layers via
// V_WMMA_F32_16X16X32_F16, one wave computes a 16-row output tile,
// inputs staged/gathered into LDS as f32, converted to f16 fragments.
// Global-memory pointers are explicitly addrspace(1)-cast so weight and
// activation traffic lowers to global_load_* (LOADcnt only) instead of
// flat_load_* (which would also tie up DScnt against the LDS tiles).
//
// Model dims (from reference): S=14, HID=32, EMB=32, CF=32, Z=64,
// B=256, T_AB=30, K=4, HORIZON=149.

#define KOBJ 4

typedef __attribute__((ext_vector_type(16))) _Float16 v16h;
typedef __attribute__((ext_vector_type(8)))  float    v8f;

// Explicit global-address-space pointers (generic -> AS1 downcast).
typedef const float __attribute__((address_space(1))) * gcptr;
typedef float __attribute__((address_space(1))) * gptr;

#define DEV static __device__ __forceinline__

DEV gcptr GC(const float* p) { return (gcptr)p; }
DEV gptr  GM(float* p)       { return (gptr)p; }

// K-index layout for 16-bit A (16x32) / B (32x16) fragments, wave32:
// lanes 0-15: K = {0..7, 16..23}; lanes 16-31: K = {8..15, 24..31}.
DEV int kmap16(int e, int half) { return (e & 7) | ((e >> 3) << 4) | (half << 3); }

// A fragment from an LDS-staged f32 tile [16][kpad], kpad multiple of 32,
// zero-padded past the true K dim. (Pointer stays generic: it's LDS.)
DEV v16h load_a_lds(const float* t, int kpad, int k0) {
  int lane = (int)(threadIdx.x & 31u);
  int row = lane & 15, half = lane >> 4;
  v16h a;
#pragma unroll
  for (int e = 0; e < 16; ++e) {
    int k = k0 + kmap16(e, half);
    a[e] = (_Float16)t[row * kpad + k];
  }
  return a;
}

// B fragment from a global f32 weight matrix.
// TRANS=false: B(k,n) = W[k*ldw + n]   (x @ W, W is [K][N])
// TRANS=true : B(k,n) = W[n*ldw + k]   (x @ W.T, W is [N][K], GRU wih/whh)
template<bool TRANS>
DEV v16h load_b_w(const float* W, int ldw, int KD, int ND, int k0, int n0) {
  gcptr Wg = GC(W);
  int lane = (int)(threadIdx.x & 31u);
  int n = n0 + (lane & 15), half = lane >> 4;
  v16h b;
#pragma unroll
  for (int e = 0; e < 16; ++e) {
    int k = k0 + kmap16(e, half);
    float v = 0.f;
    if (k < KD && n < ND) v = TRANS ? Wg[(long)n * ldw + k] : Wg[(long)k * ldw + n];
    b[e] = (_Float16)v;
  }
  return b;
}

DEV v8f wmma_f16(v16h a, v16h b, v8f c) {
  // (neg_a, A, neg_b, B, c_mod, C, reuse_a, reuse_b)
  return __builtin_amdgcn_wmma_f32_16x16x32_f16(false, a, false, b, (short)0, c,
                                                false, false);
}

// One dense layer on a 16-row tile: out[16][ND] = act(in[16][KD] @ W + b).
// in: LDS tile [16][kpad]; out: LDS tile with leading dim ldo.
// D-fragment layout: value (M = r + 8*(lane>=16), N = lane&15) in acc[r].
template<int KD, int ND, bool RELU, bool TRANSB>
DEV void dense16(const float* in, int kpad, const float* W, int ldw,
                 const float* bias, float* out, int ldo) {
  int lane = (int)(threadIdx.x & 31u);
  int half = lane >> 4;
#pragma unroll
  for (int n0 = 0; n0 < ND; n0 += 16) {
    v8f acc = {0.f, 0.f, 0.f, 0.f, 0.f, 0.f, 0.f, 0.f};
    for (int k0 = 0; k0 < KD; k0 += 32) {
      v16h a = load_a_lds(in, kpad, k0);
      v16h b = load_b_w<TRANSB>(W, ldw, KD, ND, k0, n0);
      acc = wmma_f16(a, b, acc);
    }
    int n = n0 + (lane & 15);
#pragma unroll
    for (int r = 0; r < 8; ++r) {
      int mrow = r + 8 * half;
      float v = acc[r];
      if (bias) v += GC(bias)[n < ND ? n : 0];
      if (RELU) v = v > 0.f ? v : 0.f;
      if (n < ND) out[mrow * ldo + n] = v;
    }
  }
}

// ---------------------------------------------------------------------------
// Edge MLP of an interaction-net GCN over fully-connected K=4 graphs.
// Edge row ((n*4+i)*4 + j): input = concat(x[n,i,:], x[n,j,:]) (2*SIN dims).
// Per-node feature gather: f < split -> X1[node*s1+f], else X2[node*s2+f-split].
// NL chained [.. -> 32] relu layers; writes e[row][32].
template<int SIN, int NL>
__global__ void __launch_bounds__(32)
edge_mlp(const float* __restrict__ X1, long s1,
         const float* __restrict__ X2, long s2, int split,
         const float* W0, const float* B0,
         const float* W1, const float* B1,
         const float* W2, const float* B2,
         float* __restrict__ eout, long nedges) {
  constexpr int KD0 = 2 * SIN;
  constexpr int KP0 = ((KD0 + 31) / 32) * 32;
  __shared__ float s_in[16 * KP0];
  __shared__ float s_h0[16 * 32];
  __shared__ float s_h1[16 * 32];
  gcptr x1 = GC(X1); gcptr x2 = GC(X2);
  int lane = (int)threadIdx.x;
  long row0 = (long)blockIdx.x * 16;
  for (int idx = lane; idx < 16 * KP0; idx += 32) {
    int r = idx / KP0, f = idx % KP0;
    long row = row0 + r; if (row >= nedges) row = nedges - 1;
    long n = row >> 4;
    int ij = (int)(row & 15);
    int i = ij >> 2, j = ij & 3;
    float v = 0.f;
    if (f < KD0) {
      int node = (f < SIN) ? i : j;
      int fl = (f < SIN) ? f : f - SIN;
      long nr = n * KOBJ + node;
      v = (fl < split) ? x1[nr * s1 + fl] : x2[nr * s2 + (fl - split)];
    }
    s_in[r * KP0 + f] = v;
  }
  __syncthreads();
  dense16<KD0, 32, true, false>(s_in, KP0, W0, 32, B0, s_h0, 32);
  __syncthreads();
  if (NL >= 2) { dense16<32, 32, true, false>(s_h0, 32, W1, 32, B1, s_h1, 32); __syncthreads(); }
  if (NL >= 3) { dense16<32, 32, true, false>(s_h1, 32, W2, 32, B2, s_h0, 32); __syncthreads(); }
  const float* res = (NL == 2) ? s_h1 : s_h0;
  gptr eo = GM(eout);
  for (int idx = lane; idx < 16 * 32; idx += 32) {
    int r = idx >> 5, cc = idx & 31;
    long row = row0 + r;
    if (row < nedges) eo[row * 32 + cc] = res[r * 32 + cc];
  }
}

// m[node][c] = sum_j e[(node*4+j)][c]
__global__ void reduce_m(const float* __restrict__ e, float* __restrict__ m, long nnodes) {
  long idx = (long)blockIdx.x * blockDim.x + threadIdx.x;
  if (idx >= nnodes * 32) return;
  long node = idx >> 5; int cc = (int)(idx & 31);
  gcptr eg = GC(e);
  float s = 0.f;
  for (int j = 0; j < KOBJ; ++j) s += eg[(node * KOBJ + j) * 32 + cc];
  GM(m)[node * 32 + cc] = s;
}

// Node MLP: input concat(x[node][:SIN], m[node][:32]); relu(K->32) then 32->HOUT.
// omode 0: out[row*ostride + c].  omode 1 (conf reorder [B*T,K]->[B*K,T]): T given.
template<int SIN, int HOUT>
__global__ void __launch_bounds__(32)
node_mlp(const float* __restrict__ X1, long s1,
         const float* __restrict__ X2, long s2, int split,
         const float* __restrict__ m,
         const float* W0, const float* B0,
         const float* W1, const float* B1,
         float* __restrict__ out, long ostride, int omode, int T, long nrows) {
  constexpr int KD = SIN + 32;
  constexpr int KP = ((KD + 31) / 32) * 32;
  constexpr int HP = ((HOUT + 15) / 16) * 16;
  __shared__ float s_in[16 * KP];
  __shared__ float s_h[16 * 32];
  __shared__ float s_o[16 * HP];
  gcptr x1 = GC(X1); gcptr x2 = GC(X2); gcptr mg = GC(m);
  int lane = (int)threadIdx.x;
  long row0 = (long)blockIdx.x * 16;
  for (int idx = lane; idx < 16 * KP; idx += 32) {
    int r = idx / KP, f = idx % KP;
    long row = row0 + r; if (row >= nrows) row = nrows - 1;
    float v = 0.f;
    if (f < split)      v = x1[row * s1 + f];
    else if (f < SIN)   v = x2[row * s2 + (f - split)];
    else if (f < KD)    v = mg[row * 32 + (f - SIN)];
    s_in[r * KP + f] = v;
  }
  __syncthreads();
  dense16<KD, 32, true, false>(s_in, KP, W0, 32, B0, s_h, 32);
  __syncthreads();
  dense16<32, HOUT, false, false>(s_h, 32, W1, HOUT, B1, s_o, HP);
  __syncthreads();
  gptr og = GM(out);
  for (int idx = lane; idx < 16 * HOUT; idx += 32) {
    int r = idx / HOUT, cc = idx % HOUT;
    long row = row0 + r;
    if (row < nrows) {
      long o;
      if (omode == 0) o = row * ostride + cc;
      else {
        long n = row >> 2; int i = (int)(row & 3);
        long b = n / T; int t = (int)(n % T);
        o = ((b * KOBJ + i) * (long)T + t) * (long)HOUT + cc;
      }
      og[o] = s_o[r * HP + cc];
    }
  }
}

// Full-sequence GRU (torch cell math), input dim == hidden dim == H.
// x: [nseq][T][H]; out_last=1 -> out[seq][H] (final step), else out[seq][T][H].
template<int NL, int H>
__global__ void __launch_bounds__(32)
gru_seq(const float* __restrict__ x,
        const float* wih0, const float* whh0, const float* bih0, const float* bhh0,
        const float* wih1, const float* whh1, const float* bih1, const float* bhh1,
        int T, float* __restrict__ out, int out_last) {
  constexpr int G = 3 * H;
  __shared__ float s_x[16 * H];
  __shared__ float s_h[NL][16 * H];
  __shared__ float s_gi[16 * G];
  __shared__ float s_gh[16 * G];
  int lane = (int)threadIdx.x;
  long seq0 = (long)blockIdx.x * 16;
  gcptr xg = GC(x);
  gptr og = GM(out);
  const float* wih[2] = { wih0, wih1 };
  const float* whh[2] = { whh0, whh1 };
  const float* bih[2] = { bih0, bih1 };
  const float* bhh[2] = { bhh0, bhh1 };
  for (int idx = lane; idx < NL * 16 * H; idx += 32) (&s_h[0][0])[idx] = 0.f;
  __syncthreads();
  for (int t = 0; t < T; ++t) {
    for (int idx = lane; idx < 16 * H; idx += 32) {
      int r = idx / H, f = idx % H;
      s_x[idx] = xg[((seq0 + r) * (long)T + t) * H + f];
    }
    __syncthreads();
    const float* inp = s_x;
    for (int l = 0; l < NL; ++l) {
      dense16<H, G, false, true>(inp, H, wih[l], H, nullptr, s_gi, G);
      dense16<H, G, false, true>(&s_h[l][0], H, whh[l], H, nullptr, s_gh, G);
      __syncthreads();
      gcptr bi = GC(bih[l]); gcptr bh = GC(bhh[l]);
      for (int idx = lane; idx < 16 * H; idx += 32) {
        int r = idx / H, u = idx % H;
        float ir  = s_gi[r * G + u]         + bi[u];
        float iz  = s_gi[r * G + H + u]     + bi[H + u];
        float in_ = s_gi[r * G + 2 * H + u] + bi[2 * H + u];
        float hr  = s_gh[r * G + u]         + bh[u];
        float hz  = s_gh[r * G + H + u]     + bh[H + u];
        float hn  = s_gh[r * G + 2 * H + u] + bh[2 * H + u];
        float rg = 1.f / (1.f + __expf(-(ir + hr)));
        float zg = 1.f / (1.f + __expf(-(iz + hz)));
        float ng = tanhf(in_ + rg * hn);
        float ho = s_h[l][idx];
        s_h[l][idx] = (1.f - zg) * ng + zg * ho;
      }
      __syncthreads();
      inp = &s_h[l][0];
    }
    if (!out_last) {
      for (int idx = lane; idx < 16 * H; idx += 32) {
        int r = idx / H, u = idx % H;
        og[((seq0 + r) * (long)T + t) * H + u] = s_h[NL - 1][idx];
      }
    }
    __syncthreads();
  }
  if (out_last) {
    for (int idx = lane; idx < 16 * H; idx += 32) {
      int r = idx / H, u = idx % H;
      og[(seq0 + r) * H + u] = s_h[NL - 1][idx];
    }
  }
}

// Single GRU step, 2 layers, H=32, state persists in global hstate[2][nseq][32].
__global__ void __launch_bounds__(32)
gru_step2(const float* __restrict__ x, float* __restrict__ hstate,
          const float* wih0, const float* whh0, const float* bih0, const float* bhh0,
          const float* wih1, const float* whh1, const float* bih1, const float* bhh1,
          long nseq) {
  constexpr int H = 32, G = 96;
  __shared__ float s_x[16 * H];
  __shared__ float s_h[16 * H];
  __shared__ float s_gi[16 * G];
  __shared__ float s_gh[16 * G];
  int lane = (int)threadIdx.x;
  long row0 = (long)blockIdx.x * 16;
  const float* wih[2] = { wih0, wih1 };
  const float* whh[2] = { whh0, whh1 };
  const float* bih[2] = { bih0, bih1 };
  const float* bhh[2] = { bhh0, bhh1 };
  gcptr xg = GC(x);
  for (int idx = lane; idx < 16 * H; idx += 32) {
    int r = idx / H, f = idx % H;
    s_x[idx] = xg[(row0 + r) * H + f];
  }
  for (int l = 0; l < 2; ++l) {
    gptr hg = GM(hstate) + (long)l * nseq * H;
    for (int idx = lane; idx < 16 * H; idx += 32) {
      int r = idx / H, f = idx % H;
      s_h[idx] = hg[(row0 + r) * H + f];
    }
    __syncthreads();
    dense16<H, G, false, true>(s_x, H, wih[l], H, nullptr, s_gi, G);
    dense16<H, G, false, true>(s_h, H, whh[l], H, nullptr, s_gh, G);
    __syncthreads();
    gcptr bi = GC(bih[l]); gcptr bh = GC(bhh[l]);
    for (int idx = lane; idx < 16 * H; idx += 32) {
      int r = idx / H, u = idx % H;
      float ir  = s_gi[r * G + u]         + bi[u];
      float iz  = s_gi[r * G + H + u]     + bi[H + u];
      float in_ = s_gi[r * G + 2 * H + u] + bi[2 * H + u];
      float hr  = s_gh[r * G + u]         + bh[u];
      float hz  = s_gh[r * G + H + u]     + bh[H + u];
      float hn  = s_gh[r * G + 2 * H + u] + bh[2 * H + u];
      float rg = 1.f / (1.f + __expf(-(ir + hr)));
      float zg = 1.f / (1.f + __expf(-(iz + hz)));
      float ng = tanhf(in_ + rg * hn);
      float hnew = (1.f - zg) * ng + zg * s_h[idx];
      hg[(row0 + r) * H + u] = hnew;
      s_x[idx] = hnew;                 // input to next layer
    }
    __syncthreads();
  }
}

// z_next = z_cur + h_top @ Wlin (32->64), strided rows (stride in floats).
__global__ void __launch_bounds__(32)
zupdate(const float* __restrict__ htop, const float* __restrict__ Wlin,
        const float* __restrict__ zcur, float* __restrict__ znext, long rstride) {
  __shared__ float s_in[16 * 32];
  __shared__ float s_o[16 * 64];
  int lane = (int)threadIdx.x;
  long row0 = (long)blockIdx.x * 16;
  gcptr hg = GC(htop);
  for (int idx = lane; idx < 16 * 32; idx += 32) {
    int r = idx >> 5, f = idx & 31;
    s_in[idx] = hg[(row0 + r) * 32 + f];
  }
  __syncthreads();
  dense16<32, 64, false, false>(s_in, 32, Wlin, 64, nullptr, s_o, 64);
  __syncthreads();
  gcptr zc = GC(zcur); gptr zn = GM(znext);
  for (int idx = lane; idx < 16 * 64; idx += 32) {
    int r = idx >> 6, cc = idx & 63;
    zn[(row0 + r) * rstride + cc] = zc[(row0 + r) * rstride + cc] + s_o[idx];
  }
}

// zd [b*4+i][t][64] -> zdp [(b*T+t)*4+i][64]
__global__ void permute_zd(const float* __restrict__ in, float* __restrict__ out, int T) {
  long idx = (long)blockIdx.x * blockDim.x + threadIdx.x;
  long total = 1024L * T * 64;
  if (idx >= total) return;
  long cc = idx & 63;
  long rest = idx >> 6;
  long t = rest % T;
  long sk = rest / T;
  long b = sk >> 2; long i = sk & 3;
  GM(out)[((b * T + t) * KOBJ + i) * 64 + cc] = GC(in)[idx];
}

__global__ void zero_f(float* __restrict__ p, long n) {
  long i = (long)blockIdx.x * blockDim.x + threadIdx.x;
  if (i < n) GM(p)[i] = 0.f;
}

// ---------------------------------------------------------------------------
extern "C" void kernel_launch(void* const* d_in, const int* in_sizes, int n_in,
                              void* d_out, int out_size, void* d_ws, size_t ws_size,
                              hipStream_t stream) {
  (void)in_sizes; (void)n_in; (void)out_size; (void)ws_size;
  // Assumed flattening: setup_inputs() insertion order, params pytree
  // depth-first in insertion order, lin = {w,b}, gru layer = {wih,whh,bih,bhh}.
  //  0 ab [256,30,4,14]   1 c [256,4,14]
  //  conf_gcn edge: 2w 3b / 4w 5b / 6w 7b   node: 8w 9b / 10w 11b
  //  conf_gru L0: 12 wih 13 whh 14 bih 15 bhh   L1: 16..19
  //  enc_gcn  edge: 20w 21b   node: 22w 23b / 24w 25b
  //  dyn_gcn  edge: 26w 27b / 28w 29b / 30w 31b   node: 32w 33b / 34w 35b
  //  dyn_gru  L0: 36..39   L1: 40..43
  //  dyn_lin_w: 44
  //  dec_gru  L0: 45..48
  //  dec_gcn  edge: 49w 50b   node: 51w 52b / 53w 54b
  //  55: horizon (unused; hardcoded 149)
  const float* P[64];
  for (int i = 0; i < n_in && i < 64; ++i) P[i] = (const float*)d_in[i];
  const float* ab = P[0];
  const float* cc = P[1];
  float* out = (float*)d_out;

  // Workspace layout (floats)
  float* ws    = (float*)d_ws;
  float* e_buf = ws;                     // 614400*32 = 19,660,800
  float* m_buf = e_buf + 19660800L;      // 153600*32 =  4,915,200
  float* emb   = m_buf + 4915200L;       // 1024*30*32 =   983,040
  float* cf    = emb + 983040L;          // 1024*32    =    32,768
  float* zs    = cf + 32768L;            // 1024*150*64 = 9,830,400
  float* enode = zs + 9830400L;          // 1024*32    =    32,768
  float* hst   = enode + 32768L;         // 2*1024*32  =    65,536
  float* zd    = hst + 65536L;           // 1024*150*64 = 9,830,400
  float* zdp   = zd + 9830400L;          // 1024*150*64 = 9,830,400

  const long ZROW = 150L * 64;           // per-seq row stride inside zs

  // ---- 1. Confounders GCN: 7680 graphs -> emb [1024][30][32] (reordered) ----
  edge_mlp<14, 3><<<7680, 32, 0, stream>>>(ab, 14, ab, 14, 14,
      P[2], P[3], P[4], P[5], P[6], P[7], e_buf, 122880L);
  reduce_m<<<(int)((30720L * 32 + 255) / 256), 256, 0, stream>>>(e_buf, m_buf, 30720L);
  node_mlp<14, 32><<<1920, 32, 0, stream>>>(ab, 14, ab, 14, 14, m_buf,
      P[8], P[9], P[10], P[11], emb, 32, /*omode*/1, /*T*/30, 30720L);

  // ---- 2. Confounders GRU (2 layers, H=32, T=30, last step) -> cf [1024][32]
  gru_seq<2, 32><<<64, 32, 0, stream>>>(emb,
      P[12], P[13], P[14], P[15], P[16], P[17], P[18], P[19],
      30, cf, /*out_last*/1);

  // ---- 3. Encoder GCN: c -> z0 stored at zs[:, t=0, :] ----
  edge_mlp<14, 1><<<256, 32, 0, stream>>>(cc, 14, cc, 14, 14,
      P[20], P[21], nullptr, nullptr, nullptr, nullptr, e_buf, 4096L);
  reduce_m<<<(int)((1024L * 32 + 255) / 256), 256, 0, stream>>>(e_buf, m_buf, 1024L);
  node_mlp<14, 64><<<64, 32, 0, stream>>>(cc, 14, cc, 14, 14, m_buf,
      P[22], P[23], P[24], P[25], zs, ZROW, 0, 1, 1024L);

  // ---- 4. Dynamic rollout (149 sequential steps) ----
  zero_f<<<(int)((2L * 1024 * 32 + 255) / 256), 256, 0, stream>>>(hst, 2L * 1024 * 32);
  for (int t = 0; t < 149; ++t) {
    const float* zcur = zs + (long)t * 64;   // row stride ZROW
    edge_mlp<96, 3><<<256, 32, 0, stream>>>(zcur, ZROW, cf, 32, 64,
        P[26], P[27], P[28], P[29], P[30], P[31], e_buf, 4096L);
    reduce_m<<<(int)((1024L * 32 + 255) / 256), 256, 0, stream>>>(e_buf, m_buf, 1024L);
    node_mlp<96, 32><<<64, 32, 0, stream>>>(zcur, ZROW, cf, 32, 64, m_buf,
        P[32], P[33], P[34], P[35], enode, 32, 0, 1, 1024L);
    gru_step2<<<64, 32, 0, stream>>>(enode, hst,
        P[36], P[37], P[38], P[39], P[40], P[41], P[42], P[43], 1024L);
    zupdate<<<64, 32, 0, stream>>>(hst + 1024L * 32, P[44],
        zcur, zs + (long)(t + 1) * 64, ZROW);
  }

  // ---- 5. Decoder GRU (1 layer, H=64, T=150) over zs -> zd [1024][150][64] ----
  gru_seq<1, 64><<<64, 32, 0, stream>>>(zs,
      P[45], P[46], P[47], P[48], P[45], P[46], P[47], P[48],
      150, zd, /*out_last*/0);
  permute_zd<<<(int)((9830400L + 255) / 256), 256, 0, stream>>>(zd, zdp, 150);

  // ---- 6. Decoder GCN: 38400 graphs -> out [256,150,4,14] ----
  edge_mlp<64, 1><<<38400, 32, 0, stream>>>(zdp, 64, zdp, 64, 64,
      P[49], P[50], nullptr, nullptr, nullptr, nullptr, e_buf, 614400L);
  reduce_m<<<(int)((153600L * 32 + 255) / 256), 256, 0, stream>>>(e_buf, m_buf, 153600L);
  node_mlp<64, 14><<<9600, 32, 0, stream>>>(zdp, 64, zdp, 64, 64, m_buf,
      P[51], P[52], P[53], P[54], out, 14, 0, 1, 153600L);
}
